// MultiHashTimeRadianceField_47141561041217
// MI455X (gfx1250) — matrix operand order repaired and checked by reading, not compiled, not run on hardware
//
#include <hip/hip_runtime.h>
#include <hip/hip_bf16.h>
#include <math.h>

#define NLEV 16
#define TSIZE (1 << 19)

typedef __attribute__((ext_vector_type(16))) _Float16 v16h;
typedef __attribute__((ext_vector_type(8)))  _Float16 v8h;
typedef __attribute__((ext_vector_type(8)))  float    v8f;

struct LevelParams {
    float scale[NLEV];
    int   res[NLEV];
    int   dense[NLEV];
};

// A-fragment (16x32 f16): lane (m=l%16, half=l/16) holds row m,
// K = colbase..colbase+7 (v0..v3) and colbase+16..colbase+23 (v4..v7),
// where colbase = 32*kt + 8*half.
__device__ __forceinline__ v16h load_a_frag(const _Float16* row, int colbase) {
    v8h lo = *(const v8h*)(row + colbase);
    v8h hi = *(const v8h*)(row + colbase + 16);
    v16h a;
#pragma unroll
    for (int i = 0; i < 8; ++i) { a[i] = lo[i]; a[8 + i] = hi[i]; }
    return a;
}

// B-fragment (32x16 f16): lane (n=l%16, kset=l/16) holds column n,
// K = 32*kt + 8*kset + 0..7 (v0..v3) and +16..+23 (v4..v7), kbase = 32*kt + 16*kset.
// Weights stored transposed in LDS: Wt[n][k], row stride = K-total.
__device__ __forceinline__ v16h load_b_frag(const _Float16* wt, int stride, int n, int kbase) {
    const _Float16* p = wt + n * stride + kbase - ((kbase & 16) >> 1);
    v8h lo = *(const v8h*)p;
    v8h hi = *(const v8h*)(p + 16);
    v16h b;
#pragma unroll
    for (int i = 0; i < 8; ++i) { b[i] = lo[i]; b[8 + i] = hi[i]; }
    return b;
}

__device__ __forceinline__ v8f wmma16(v16h a, v16h b, v8f c) {
    return __builtin_amdgcn_wmma_f32_16x16x32_f16(false, a, false, b, (short)0, c, false, false);
}

__global__ __launch_bounds__(256)
void ngp_field_kernel(const float* __restrict__ xyzs, const float* __restrict__ dirs,
                      const float* __restrict__ ta1, const float* __restrict__ ta2,
                      const float* __restrict__ tb1, const float* __restrict__ tb2,
                      const float* __restrict__ tst,
                      const float* __restrict__ w1,  const float* __restrict__ w2,
                      const float* __restrict__ wc1, const float* __restrict__ wc2,
                      const float* __restrict__ wc3,
                      float* __restrict__ out, int n, LevelParams lp)
{
    // ---- LDS: f16 transposed weights (28 KB) + per-wave staging (24 KB) ----
    __shared__ _Float16 sW1t [64 * 96];  // layer xyz1: Wt[n=0..63][k=0..95]
    __shared__ _Float16 sW2t [16 * 64];  // layer xyz2: Wt[n=0..15][k=0..63]
    __shared__ _Float16 sWc1t[64 * 32];  // color1:     Wt[n=0..63][k=0..31]
    __shared__ _Float16 sWc2t[64 * 64];  // color2:     Wt[n=0..63][k=0..63]
    __shared__ _Float16 sWc3t[16 * 64];  // color3 padded to 16 cols: Wt[n][k]
    __shared__ _Float16 sA[8][16 * 96];  // per-wave activation staging, row stride 96

    const int tid = threadIdx.x;

    // ---- convert weights f32 -> f16, transposed ----
    for (int i = tid; i < 64 * 96; i += 256) { int nn = i / 96, kk = i % 96; sW1t[i]  = (_Float16)w1 [kk * 64 + nn]; }
    for (int i = tid; i < 16 * 64; i += 256) { int nn = i / 64, kk = i % 64; sW2t[i]  = (_Float16)w2 [kk * 16 + nn]; }
    for (int i = tid; i < 64 * 32; i += 256) { int nn = i / 32, kk = i % 32; sWc1t[i] = (_Float16)wc1[kk * 64 + nn]; }
    for (int i = tid; i < 64 * 64; i += 256) { int nn = i / 64, kk = i % 64; sWc2t[i] = (_Float16)wc2[kk * 64 + nn]; }
    for (int i = tid; i < 16 * 64; i += 256) { int nn = i / 64, kk = i % 64;
        sWc3t[i] = (nn < 3) ? (_Float16)wc3[kk * 3 + nn] : (_Float16)0.0f; }
    __syncthreads();

    const int wave = tid >> 5;
    const int lane = tid & 31;
    const int m    = lane & 15;   // point-in-tile / A row / C column
    const int half = lane >> 4;   // A K-chunk selector / B K-set / C row-half

    const int base = (blockIdx.x * 8 + wave) * 16;
    if (base >= n) return;
    int p = base + m; if (p >= n) p = n - 1;

    _Float16* Arow = &sA[wave][0];

    // ---------------- phase 1: multiresolution hash encode ----------------
    // lane (m, half) encodes point p for levels 8*half .. 8*half+7.
    const float tval  = xyzs[3];
    float f1 = tval * 16.0f; const float next1 = f1 - floorf(f1); const float prev1 = 1.0f - next1;
    float f2 = tval * 20.0f; const float next2 = f2 - floorf(f2); const float prev2 = 1.0f - next2;

    const float px = (xyzs[(size_t)p * 4 + 0] + 1.0f) * 0.5f;
    const float py = (xyzs[(size_t)p * 4 + 1] + 1.0f) * 0.5f;
    const float pz = (xyzs[(size_t)p * 4 + 2] + 1.0f) * 0.5f;

    for (int li = 0; li < 8; ++li) {
        const int   lev   = half * 8 + li;
        const float s     = lp.scale[lev];
        const int   res   = lp.res[lev];
        const int   dense = lp.dense[lev];

        const float fx = px * s + 0.5f, fy = py * s + 0.5f, fz = pz * s + 0.5f;
        const float gx = floorf(fx), gy = floorf(fy), gz = floorf(fz);
        const float wx = fx - gx, wy = fy - gy, wz = fz - gz;
        const int ix = (int)gx, iy = (int)gy, iz = (int)gz;

        float a1x=0.f,a1y=0.f,a2x=0.f,a2y=0.f,b1x=0.f,b1y=0.f,b2x=0.f,b2y=0.f,stx=0.f,sty=0.f;
        const size_t lbase = (size_t)lev * (TSIZE * 2);

        for (int c = 0; c < 8; ++c) {
            const int cx = ix + (c & 1), cy = iy + ((c >> 1) & 1), cz = iz + ((c >> 2) & 1);
            unsigned idx;
            if (dense) idx = (unsigned)(cx + cy * res + cz * res * res);
            else       idx = ((unsigned)cx * 1u) ^ ((unsigned)cy * 2654435761u) ^ ((unsigned)cz * 805459861u);
            idx &= (unsigned)(TSIZE - 1);
            const float wt = ((c & 1) ? wx : 1.0f - wx) *
                             ((c & 2) ? wy : 1.0f - wy) *
                             ((c & 4) ? wz : 1.0f - wz);
            const size_t off = lbase + (size_t)idx * 2;
            float2 v;
            v = *(const float2*)(ta1 + off); a1x += wt * v.x; a1y += wt * v.y;
            v = *(const float2*)(ta2 + off); a2x += wt * v.x; a2y += wt * v.y;
            v = *(const float2*)(tb1 + off); b1x += wt * v.x; b1y += wt * v.y;
            v = *(const float2*)(tb2 + off); b2x += wt * v.x; b2y += wt * v.y;
            v = *(const float2*)(tst + off); stx += wt * v.x; sty += wt * v.y;
        }

        // time-blended features -> staging row (cols: blend_a 0-31, blend_b 32-63, static 64-95)
        _Float16* r = Arow + m * 96;
        r[2 * lev]          = (_Float16)(a1x * prev1 + a2x * next1);
        r[2 * lev + 1]      = (_Float16)(a1y * prev1 + a2y * next1);
        r[32 + 2 * lev]     = (_Float16)(b1x * prev2 + b2x * next2);
        r[32 + 2 * lev + 1] = (_Float16)(b1y * prev2 + b2y * next2);
        r[64 + 2 * lev]     = (_Float16)stx;
        r[64 + 2 * lev + 1] = (_Float16)sty;

        if (lev >= 12) { // feature1/feature2 outputs (last 8 of fa/fb)
            const int j = lev - 12;
            float* fo1 = out + (size_t)4  * n + (size_t)p * 16;
            float* fo2 = out + (size_t)20 * n + (size_t)p * 16;
            fo1[2 * j] = a1x; fo1[2 * j + 1] = a1y; fo1[8 + 2 * j] = b1x; fo1[8 + 2 * j + 1] = b1y;
            fo2[2 * j] = a2x; fo2[2 * j + 1] = a2y; fo2[8 + 2 * j] = b2x; fo2[8 + 2 * j + 1] = b2y;
        }
    }

    const _Float16* arow = Arow + m * 96;
    const int nl   = m;     // output-column role of this lane
    const int kset = half;  // B K-set role of this lane

    // ---------------- layer 1: [16,96] @ [96,64], relu ----------------
    {
        v16h a0 = load_a_frag(arow,  0 + half * 8);
        v16h a1 = load_a_frag(arow, 32 + half * 8);
        v16h a2 = load_a_frag(arow, 64 + half * 8);
#pragma unroll
        for (int nt = 0; nt < 4; ++nt) {
            v8f acc = {0.f,0.f,0.f,0.f,0.f,0.f,0.f,0.f};
            acc = wmma16(a0, load_b_frag(sW1t, 96, nt * 16 + nl,  0 + kset * 16), acc);
            acc = wmma16(a1, load_b_frag(sW1t, 96, nt * 16 + nl, 32 + kset * 16), acc);
            acc = wmma16(a2, load_b_frag(sW1t, 96, nt * 16 + nl, 64 + kset * 16), acc);
#pragma unroll
            for (int r = 0; r < 8; ++r) {
                float v = acc[r] > 0.0f ? acc[r] : 0.0f;
                Arow[(r + 8 * half) * 96 + nt * 16 + nl] = (_Float16)v;
            }
        }
    }

    // ---------------- layer 2: h = [16,64] @ [64,16] (no relu) ----------------
    {
        v16h a0 = load_a_frag(arow,  0 + half * 8);
        v16h a1 = load_a_frag(arow, 32 + half * 8);
        v8f acc = {0.f,0.f,0.f,0.f,0.f,0.f,0.f,0.f};
        acc = wmma16(a0, load_b_frag(sW2t, 64, nl,  0 + kset * 16), acc);
        acc = wmma16(a1, load_b_frag(sW2t, 64, nl, 32 + kset * 16), acc);
        // sigma = exp(h[:,0]) -> lanes with nl==0 own column 0
        if (nl == 0) {
#pragma unroll
            for (int r = 0; r < 8; ++r) {
                const int q = base + r + 8 * half;
                if (q < n) out[(size_t)3 * n + q] = expf(acc[r]);
            }
        }
        // stage h into ch cols 16-31
#pragma unroll
        for (int r = 0; r < 8; ++r)
            Arow[(r + 8 * half) * 96 + 16 + nl] = (_Float16)acc[r];
    }

    // ---------------- SH degree-4 dir encoding -> ch cols 0-15 ----------------
    if (half == 0) {
        const float x = dirs[(size_t)p * 3 + 0] * 2.0f - 1.0f;
        const float y = dirs[(size_t)p * 3 + 1] * 2.0f - 1.0f;
        const float z = dirs[(size_t)p * 3 + 2] * 2.0f - 1.0f;
        const float xy = x * y, xz = x * z, yz = y * z;
        const float x2 = x * x, y2 = y * y, z2 = z * z;
        _Float16* r = Arow + m * 96;
        r[0]  = (_Float16)0.28209479177387814f;
        r[1]  = (_Float16)(-0.48860251190291987f * y);
        r[2]  = (_Float16)( 0.48860251190291987f * z);
        r[3]  = (_Float16)(-0.48860251190291987f * x);
        r[4]  = (_Float16)( 1.0925484305920792f * xy);
        r[5]  = (_Float16)(-1.0925484305920792f * yz);
        r[6]  = (_Float16)( 0.94617469575756f * z2 - 0.31539156525252005f);
        r[7]  = (_Float16)(-1.0925484305920792f * xz);
        r[8]  = (_Float16)( 0.5462742152960396f * (x2 - y2));
        r[9]  = (_Float16)( 0.5900435899266435f * y * (-3.0f * x2 + y2));
        r[10] = (_Float16)( 2.890611442640554f * xy * z);
        r[11] = (_Float16)( 0.4570457994644657f * y * (1.0f - 5.0f * z2));
        r[12] = (_Float16)( 0.3731763325901154f * z * (5.0f * z2 - 3.0f));
        r[13] = (_Float16)( 0.4570457994644657f * x * (1.0f - 5.0f * z2));
        r[14] = (_Float16)( 1.445305721320277f * z * (x2 - y2));
        r[15] = (_Float16)( 0.5900435899266435f * x * (-x2 + 3.0f * y2));
    }

    // ---------------- color layer 1: [16,32] @ [32,64], relu ----------------
    {
        v16h a0 = load_a_frag(arow, 0 + half * 8); // ch cols 0-31
#pragma unroll
        for (int nt = 0; nt < 4; ++nt) {
            v8f acc = {0.f,0.f,0.f,0.f,0.f,0.f,0.f,0.f};
            acc = wmma16(a0, load_b_frag(sWc1t, 32, nt * 16 + nl, kset * 16), acc);
#pragma unroll
            for (int r = 0; r < 8; ++r) {
                float v = acc[r] > 0.0f ? acc[r] : 0.0f;
                Arow[(r + 8 * half) * 96 + nt * 16 + nl] = (_Float16)v;
            }
        }
    }

    // ---------------- color layer 2: [16,64] @ [64,64], relu ----------------
    {
        v16h a0 = load_a_frag(arow,  0 + half * 8);
        v16h a1 = load_a_frag(arow, 32 + half * 8);
#pragma unroll
        for (int nt = 0; nt < 4; ++nt) {
            v8f acc = {0.f,0.f,0.f,0.f,0.f,0.f,0.f,0.f};
            acc = wmma16(a0, load_b_frag(sWc2t, 64, nt * 16 + nl,  0 + kset * 16), acc);
            acc = wmma16(a1, load_b_frag(sWc2t, 64, nt * 16 + nl, 32 + kset * 16), acc);
#pragma unroll
            for (int r = 0; r < 8; ++r) {
                float v = acc[r] > 0.0f ? acc[r] : 0.0f;
                Arow[(r + 8 * half) * 96 + nt * 16 + nl] = (_Float16)v;
            }
        }
    }

    // ---------------- color layer 3: [16,64] @ [64,16(pad)], sigmoid ----------------
    {
        v16h a0 = load_a_frag(arow,  0 + half * 8);
        v16h a1 = load_a_frag(arow, 32 + half * 8);
        v8f acc = {0.f,0.f,0.f,0.f,0.f,0.f,0.f,0.f};
        acc = wmma16(a0, load_b_frag(sWc3t, 64, nl,  0 + kset * 16), acc);
        acc = wmma16(a1, load_b_frag(sWc3t, 64, nl, 32 + kset * 16), acc);
        if (nl < 3) {
#pragma unroll
            for (int r = 0; r < 8; ++r) {
                const int q = base + r + 8 * half;
                if (q < n) out[(size_t)q * 3 + nl] = 1.0f / (1.0f + expf(-acc[r]));
            }
        }
    }
}

extern "C" void kernel_launch(void* const* d_in, const int* in_sizes, int n_in,
                              void* d_out, int out_size, void* d_ws, size_t ws_size,
                              hipStream_t stream) {
    (void)n_in; (void)out_size; (void)d_ws; (void)ws_size;
    const float* xyzs = (const float*)d_in[0];
    const float* dirs = (const float*)d_in[1];
    const float* ta1  = (const float*)d_in[2];
    const float* ta2  = (const float*)d_in[3];
    const float* tb1  = (const float*)d_in[4];
    const float* tb2  = (const float*)d_in[5];
    const float* tst  = (const float*)d_in[6];
    const float* w1   = (const float*)d_in[7];
    const float* w2   = (const float*)d_in[8];
    const float* wc1  = (const float*)d_in[9];
    const float* wc2  = (const float*)d_in[10];
    const float* wc3  = (const float*)d_in[11];

    const int n = in_sizes[0] / 4;

    LevelParams lp;
    const double B = exp(log(4096.0 / 16.0) / 15.0);
    for (int l = 0; l < NLEV; ++l) {
        const double s = 16.0 * pow(B, (double)l) - 1.0;
        lp.scale[l] = (float)s;
        const int res = (int)ceil(s) + 1;
        lp.res[l] = res;
        lp.dense[l] = ((long long)res * res * res <= (long long)TSIZE) ? 1 : 0;
    }

    const int blocks = (n + 127) / 128; // 8 waves/block, 16 points/wave
    ngp_field_kernel<<<blocks, 256, 0, stream>>>(xyzs, dirs, ta1, ta2, tb1, tb2, tst,
                                                 w1, w2, wc1, wc2, wc3,
                                                 (float*)d_out, n, lp);
}
